// MambaBlock10M_16690242912260
// MI455X (gfx1250) — 8-bit matrix operand order repaired and checked by reading, compiled, not run on hardware
//
#include <hip/hip_runtime.h>
#include <cstdint>

#define TOK    4096
#define HDIM   640
#define DIN    1280
#define DXZ    2560
#define DTRN   40
#define DTRP   48
#define DTRP2  64
#define NSTP   32
#define SEQL   2048
#define NBATCH 2
#define NST    16
#define NCHUNK 64
#define CLEN   32

typedef int v8i __attribute__((ext_vector_type(8)));

__global__ void k_wabsmean(const float* __restrict__ w, int n, float* __restrict__ alpha) {
  __shared__ float s[256];
  float acc = 0.f;
  for (int i = threadIdx.x; i < n; i += 256) acc += fabsf(w[i]);
  s[threadIdx.x] = acc; __syncthreads();
  for (int st = 128; st > 0; st >>= 1) {
    if (threadIdx.x < st) s[threadIdx.x] += s[threadIdx.x + st];
    __syncthreads();
  }
  if (threadIdx.x == 0) *alpha = fmaxf(s[0] / (float)n, 1e-10f);
}

__global__ void k_wquant(const float* __restrict__ w, const float* __restrict__ alpha,
                         int8_t* __restrict__ wq, int N, int K, int Npad) {
  int idx = blockIdx.x * 256 + threadIdx.x;
  if (idx >= Npad * K) return;
  int n = idx / K;
  float q = 0.f;
  if (n < N) q = fminf(1.f, fmaxf(-1.f, rintf(w[idx] / *alpha)));
  wq[idx] = (int8_t)(int)q;
}

__global__ void k_quant_in(const float* __restrict__ x, const float* __restrict__ norm_w,
                           const float* __restrict__ in_nw,
                           int8_t* __restrict__ xq, float* __restrict__ srow) {
  __shared__ float s[256];
  const int t = blockIdx.x, tid = threadIdx.x;
  const float* xr = x + (size_t)t * HDIM;
  float v[3];
  float ss = 0.f;
#pragma unroll
  for (int i = 0; i < 3; ++i) {
    int idx = tid + i * 256;
    float f = (idx < HDIM) ? xr[idx] : 0.f;
    v[i] = f; ss += f * f;
  }
  s[tid] = ss; __syncthreads();
  for (int st = 128; st > 0; st >>= 1) { if (tid < st) s[tid] += s[tid + st]; __syncthreads(); }
  float inv1 = rsqrtf(s[0] / (float)HDIM + 1e-6f);
  __syncthreads();
  ss = 0.f;
#pragma unroll
  for (int i = 0; i < 3; ++i) {
    int idx = tid + i * 256;
    float u = (idx < HDIM) ? v[i] * inv1 * norm_w[idx] : 0.f;
    v[i] = u; ss += u * u;
  }
  s[tid] = ss; __syncthreads();
  for (int st = 128; st > 0; st >>= 1) { if (tid < st) s[tid] += s[tid + st]; __syncthreads(); }
  float inv2 = rsqrtf(s[0] / (float)HDIM + 1e-6f);
  __syncthreads();
  float mx = 0.f;
#pragma unroll
  for (int i = 0; i < 3; ++i) {
    int idx = tid + i * 256;
    float u = (idx < HDIM) ? v[i] * inv2 * in_nw[idx] : 0.f;
    v[i] = u; mx = fmaxf(mx, fabsf(u));
  }
  s[tid] = mx; __syncthreads();
  for (int st = 128; st > 0; st >>= 1) { if (tid < st) s[tid] = fmaxf(s[tid], s[tid + st]); __syncthreads(); }
  float gamma = fmaxf(s[0], 1e-10f);
  float qsc = 127.f / gamma;
#pragma unroll
  for (int i = 0; i < 3; ++i) {
    int idx = tid + i * 256;
    if (idx < HDIM) {
      float q = fminf(127.f, fmaxf(-128.f, rintf(v[i] * qsc)));
      xq[(size_t)t * HDIM + idx] = (int8_t)(int)q;
    }
  }
  if (tid == 0) srow[t] = gamma / 127.f;
}

__global__ void k_quant_din3(const float* __restrict__ xp,
                             const float* __restrict__ nw0, int8_t* __restrict__ q0, float* __restrict__ s0,
                             const float* __restrict__ nw1, int8_t* __restrict__ q1, float* __restrict__ s1,
                             const float* __restrict__ nw2, int8_t* __restrict__ q2, float* __restrict__ s2) {
  __shared__ float s[256];
  const int t = blockIdx.x, tid = threadIdx.x;
  const float* xr = xp + (size_t)t * DIN;
  float v[5]; float ss = 0.f;
#pragma unroll
  for (int i = 0; i < 5; ++i) { float f = xr[tid + i * 256]; v[i] = f; ss += f * f; }
  s[tid] = ss; __syncthreads();
  for (int st = 128; st > 0; st >>= 1) { if (tid < st) s[tid] += s[tid + st]; __syncthreads(); }
  float inv = rsqrtf(s[0] / (float)DIN + 1e-6f);
  __syncthreads();
  const float* nws[3] = { nw0, nw1, nw2 };
  int8_t* qp[3] = { q0, q1, q2 };
  float* sp[3] = { s0, s1, s2 };
  for (int p = 0; p < 3; ++p) {
    if (nws[p] == nullptr) break;
    float u[5]; float mx = 0.f;
#pragma unroll
    for (int i = 0; i < 5; ++i) {
      int idx = tid + i * 256;
      u[i] = v[i] * inv * nws[p][idx];
      mx = fmaxf(mx, fabsf(u[i]));
    }
    s[tid] = mx; __syncthreads();
    for (int st = 128; st > 0; st >>= 1) { if (tid < st) s[tid] = fmaxf(s[tid], s[tid + st]); __syncthreads(); }
    float gamma = fmaxf(s[0], 1e-10f);
    __syncthreads();
    float qsc = 127.f / gamma;
#pragma unroll
    for (int i = 0; i < 5; ++i) {
      int idx = tid + i * 256;
      float q = fminf(127.f, fmaxf(-128.f, rintf(u[i] * qsc)));
      qp[p][(size_t)t * DIN + idx] = (int8_t)(int)q;
    }
    if (tid == 0) sp[p][t] = gamma / 127.f;
  }
}

template <bool ADD_RES>
__global__ void k_gemm_iu8(const int8_t* __restrict__ Aq, const int8_t* __restrict__ Wq,
                           const float* __restrict__ srow, const float* __restrict__ alpha,
                           const float* __restrict__ resid, float* __restrict__ out,
                           int M, int Npad, int Nout, int K) {
  const int lane = threadIdx.x & 31;
  const int wave = threadIdx.x >> 5;
  const int tile = blockIdx.x * (blockDim.x >> 5) + wave;
  const int ntiles = Npad >> 5;
  const int mtiles = M >> 5;
  if (tile >= mtiles * ntiles) return;
  const int m0 = (tile / ntiles) << 5;
  const int n0 = (tile % ntiles) << 5;
  const int r = lane & 15;
  const int half = lane >> 4;
  const int8_t* Ar0 = Aq + (size_t)(m0 + r) * K;
  const int8_t* Ar1 = Aq + (size_t)(m0 + 16 + r) * K;
  const int8_t* Wr0 = Wq + (size_t)(n0 + r) * K;
  const int8_t* Wr1 = Wq + (size_t)(n0 + 16 + r) * K;
  v8i acc00 = {}, acc01 = {}, acc10 = {}, acc11 = {};
  for (int kk = 0; kk < K; kk += 64) {
    v8i a0, a1, b0, b1;
#pragma unroll
    for (int i = 0; i < 8; ++i) {
      const int ka = ((i >> 1) << 4) + ((i & 1) << 2) + (half << 3);
      const int kb = ((i >> 1) << 4) + ((i & 1) << 2) + (half << 3);
      a0[i] = *(const int*)(Ar0 + kk + ka);
      a1[i] = *(const int*)(Ar1 + kk + ka);
      b0[i] = *(const int*)(Wr0 + kk + kb);
      b1[i] = *(const int*)(Wr1 + kk + kb);
    }
    __builtin_prefetch(Ar0 + kk + 128, 0, 1);
    __builtin_prefetch(Ar1 + kk + 128, 0, 1);
    __builtin_prefetch(Wr0 + kk + 128, 0, 1);
    __builtin_prefetch(Wr1 + kk + 128, 0, 1);
    acc00 = __builtin_amdgcn_wmma_i32_16x16x64_iu8(true, a0, true, b0, acc00, false, false);
    acc01 = __builtin_amdgcn_wmma_i32_16x16x64_iu8(true, a0, true, b1, acc01, false, false);
    acc10 = __builtin_amdgcn_wmma_i32_16x16x64_iu8(true, a1, true, b0, acc10, false, false);
    acc11 = __builtin_amdgcn_wmma_i32_16x16x64_iu8(true, a1, true, b1, acc11, false, false);
  }
  const float al = *alpha;
  const v8i* accs[4] = { &acc00, &acc01, &acc10, &acc11 };
#pragma unroll
  for (int q = 0; q < 4; ++q) {
    const int mb = m0 + ((q >> 1) << 4);
    const int nb = n0 + ((q & 1) << 4);
    const int n = nb + r;
    if (n < Nout) {
      const v8i& a = *accs[q];
#pragma unroll
      for (int i = 0; i < 8; ++i) {
        const int m = mb + i + (half << 3);
        float vv = (float)a[i] * srow[m] * al;
        if (ADD_RES) vv += resid[(size_t)m * Nout + n];
        out[(size_t)m * Nout + n] = vv;
      }
    }
  }
}

__global__ void k_conv_silu(const float* __restrict__ xz, const float* __restrict__ cw,
                            const float* __restrict__ cb, float* __restrict__ xp) {
  size_t idx = (size_t)blockIdx.x * 256 + threadIdx.x;
  if (idx >= (size_t)TOK * DIN) return;
  int d = (int)(idx % DIN);
  size_t t = idx / DIN;
  int l = (int)(t % SEQL);
  float acc = cb[d];
#pragma unroll
  for (int k = 0; k < 4; ++k) {
    if (l - k >= 0) acc += cw[d * 4 + k] * xz[(t - k) * DXZ + d];
  }
  xp[idx] = acc / (1.f + expf(-acc));
}

__global__ void k_dtd(const float* __restrict__ dtu_out, const float* __restrict__ dtd_w,
                      const float* __restrict__ dtd_b, float* __restrict__ dt) {
  __shared__ float r[DTRN];
  const int t = blockIdx.x, tid = threadIdx.x;
  if (tid < DTRN) r[tid] = dtu_out[(size_t)t * DTRP + tid];
  __syncthreads();
#pragma unroll
  for (int i = 0; i < 5; ++i) {
    int d = tid + i * 256;
    float acc = dtd_b[d];
    for (int k = 0; k < DTRN; ++k) acc += r[k] * dtd_w[d * DTRN + k];
    float sp = (acc > 20.f) ? acc : log1pf(expf(acc));
    dt[(size_t)t * DIN + d] = sp;
  }
}

__global__ void k_scan_chunk(const float* __restrict__ dt, const float* __restrict__ xp,
                             const float* __restrict__ Bm, const float* __restrict__ A_log,
                             float* __restrict__ Ach, float* __restrict__ Bch) {
  const int d = blockIdx.x * 256 + threadIdx.x;
  const int c = blockIdx.y, b = blockIdx.z;
  __shared__ float Bs[NST];
  float Aneg[NST], aP[NST], bA[NST];
#pragma unroll
  for (int n = 0; n < NST; ++n) {
    Aneg[n] = -expf(A_log[d * NST + n]);
    aP[n] = 1.f; bA[n] = 0.f;
  }
  for (int s = 0; s < CLEN; ++s) {
    int l = c * CLEN + s;
    size_t tok = (size_t)b * SEQL + l;
    __syncthreads();
    if (threadIdx.x < NST) Bs[threadIdx.x] = Bm[tok * NST + threadIdx.x];
    __syncthreads();
    float dtv = dt[tok * DIN + d];
    float dtx = dtv * xp[tok * DIN + d];
#pragma unroll
    for (int n = 0; n < NST; ++n) {
      float da = __expf(Aneg[n] * dtv);
      aP[n] *= da;
      bA[n] = da * bA[n] + dtx * Bs[n];
    }
  }
  size_t o = (((size_t)b * DIN + d) * NCHUNK + c) * NST;
#pragma unroll
  for (int n = 0; n < NST; ++n) { Ach[o + n] = aP[n]; Bch[o + n] = bA[n]; }
}

__global__ void k_scan_combine(const float* __restrict__ Ach, const float* __restrict__ Bch,
                               float* __restrict__ hst) {
  int idx = blockIdx.x * 256 + threadIdx.x;
  if (idx >= NBATCH * DIN) return;
  float h[NST];
#pragma unroll
  for (int n = 0; n < NST; ++n) h[n] = 0.f;
  size_t base = (size_t)idx * NCHUNK * NST;
  for (int c = 0; c < NCHUNK; ++c) {
    size_t o = base + (size_t)c * NST;
#pragma unroll
    for (int n = 0; n < NST; ++n) hst[o + n] = h[n];
#pragma unroll
    for (int n = 0; n < NST; ++n) h[n] = Ach[o + n] * h[n] + Bch[o + n];
  }
}

__global__ void k_scan_apply(const float* __restrict__ dt, const float* __restrict__ xp,
                             const float* __restrict__ Bm, const float* __restrict__ Cm,
                             const float* __restrict__ A_log, const float* __restrict__ hst,
                             float* __restrict__ y) {
  const int d = blockIdx.x * 256 + threadIdx.x;
  const int c = blockIdx.y, b = blockIdx.z;
  __shared__ float Bs[NST], Cs[NST];
  float Aneg[NST], h[NST];
  size_t o = (((size_t)b * DIN + d) * NCHUNK + c) * NST;
#pragma unroll
  for (int n = 0; n < NST; ++n) {
    Aneg[n] = -expf(A_log[d * NST + n]);
    h[n] = hst[o + n];
  }
  for (int s = 0; s < CLEN; ++s) {
    int l = c * CLEN + s;
    size_t tok = (size_t)b * SEQL + l;
    __syncthreads();
    if (threadIdx.x < NST) Bs[threadIdx.x] = Bm[tok * NST + threadIdx.x];
    else if (threadIdx.x < 2 * NST) Cs[threadIdx.x - NST] = Cm[tok * NST + threadIdx.x - NST];
    __syncthreads();
    float dtv = dt[tok * DIN + d];
    float dtx = dtv * xp[tok * DIN + d];
    float yv = 0.f;
#pragma unroll
    for (int n = 0; n < NST; ++n) {
      float da = __expf(Aneg[n] * dtv);
      h[n] = da * h[n] + dtx * Bs[n];
      yv += h[n] * Cs[n];
    }
    y[tok * DIN + d] = yv;
  }
}

__global__ void k_combine(const float* __restrict__ xp, const float* __restrict__ Dv,
                          const float* __restrict__ xz, float* __restrict__ y) {
  size_t idx = (size_t)blockIdx.x * 256 + threadIdx.x;
  if (idx >= (size_t)TOK * DIN) return;
  int d = (int)(idx % DIN);
  size_t t = idx / DIN;
  float z = xz[t * DXZ + DIN + d];
  float yv = y[idx] + xp[idx] * Dv[d];
  y[idx] = yv * (z / (1.f + expf(-z)));
}

extern "C" void kernel_launch(void* const* d_in, const int* in_sizes, int n_in,
                              void* d_out, int out_size, void* d_ws, size_t ws_size,
                              hipStream_t stream) {
  const float* x      = (const float*)d_in[0];
  const float* norm_w = (const float*)d_in[1];
  const float* in_w   = (const float*)d_in[2];
  const float* in_nw  = (const float*)d_in[3];
  const float* conv_w = (const float*)d_in[4];
  const float* conv_b = (const float*)d_in[5];
  const float* dtu_w  = (const float*)d_in[6];
  const float* dtu_nw = (const float*)d_in[7];
  const float* dtd_w  = (const float*)d_in[8];
  const float* dtd_b  = (const float*)d_in[9];
  const float* Bp_w   = (const float*)d_in[10];
  const float* Bp_nw  = (const float*)d_in[11];
  const float* Cp_w   = (const float*)d_in[12];
  const float* Cp_nw  = (const float*)d_in[13];
  const float* A_log  = (const float*)d_in[14];
  const float* Dvec   = (const float*)d_in[15];
  const float* out_w  = (const float*)d_in[16];
  const float* out_nw = (const float*)d_in[17];
  float* out = (float*)d_out;

  char* ws = (char*)d_ws;
  size_t off = 0;
  auto alloc = [&](size_t bytes) -> char* {
    char* p = ws + off;
    off += (bytes + 255) & ~(size_t)255;
    return p;
  };
  float*  alphas = (float*)alloc(256);
  int8_t* wq_in  = (int8_t*)alloc((size_t)DXZ * HDIM);
  int8_t* wq_dtu = (int8_t*)alloc((size_t)DTRP2 * DIN);
  int8_t* wq_Bp  = (int8_t*)alloc((size_t)NSTP * DIN);
  int8_t* wq_Cp  = (int8_t*)alloc((size_t)NSTP * DIN);
  int8_t* wq_out = (int8_t*)alloc((size_t)HDIM * DIN);
  int8_t* xq_a   = (int8_t*)alloc((size_t)TOK * HDIM);
  float*  s_a    = (float*)alloc((size_t)TOK * 4);
  float*  xz     = (float*)alloc((size_t)TOK * DXZ * 4);
  float*  xpath  = (float*)alloc((size_t)TOK * DIN * 4);
  int8_t* xq_dtu = (int8_t*)alloc((size_t)TOK * DIN);
  float*  s_dtu  = (float*)alloc((size_t)TOK * 4);
  int8_t* xq_Bp  = (int8_t*)alloc((size_t)TOK * DIN);
  float*  s_Bp   = (float*)alloc((size_t)TOK * 4);
  int8_t* xq_Cp  = (int8_t*)alloc((size_t)TOK * DIN);
  float*  s_Cp   = (float*)alloc((size_t)TOK * 4);
  float*  dtuout = (float*)alloc((size_t)TOK * DTRP * 4);
  float*  dtbuf  = (float*)alloc((size_t)TOK * DIN * 4);
  float*  Bmat   = (float*)alloc((size_t)TOK * NST * 4);
  float*  Cmat   = (float*)alloc((size_t)TOK * NST * 4);
  float*  Ach    = (float*)alloc((size_t)NBATCH * DIN * NCHUNK * NST * 4);
  float*  Bch    = (float*)alloc((size_t)NBATCH * DIN * NCHUNK * NST * 4);
  float*  hst    = (float*)alloc((size_t)NBATCH * DIN * NCHUNK * NST * 4);
  float*  ybuf   = (float*)alloc((size_t)TOK * DIN * 4);
  int8_t* xq_y   = (int8_t*)alloc((size_t)TOK * DIN);
  float*  s_y    = (float*)alloc((size_t)TOK * 4);
  (void)ws_size; (void)in_sizes; (void)n_in; (void)out_size;

  k_wabsmean<<<1, 256, 0, stream>>>(in_w,  DXZ * HDIM, alphas + 0);
  k_wabsmean<<<1, 256, 0, stream>>>(dtu_w, DTRN * DIN, alphas + 1);
  k_wabsmean<<<1, 256, 0, stream>>>(Bp_w,  NST * DIN,  alphas + 2);
  k_wabsmean<<<1, 256, 0, stream>>>(Cp_w,  NST * DIN,  alphas + 3);
  k_wabsmean<<<1, 256, 0, stream>>>(out_w, HDIM * DIN, alphas + 4);
  k_wquant<<<(DXZ * HDIM + 255) / 256, 256, 0, stream>>>(in_w,  alphas + 0, wq_in,  DXZ,  HDIM, DXZ);
  k_wquant<<<(DTRP2 * DIN + 255) / 256, 256, 0, stream>>>(dtu_w, alphas + 1, wq_dtu, DTRN, DIN, DTRP2);
  k_wquant<<<(NSTP * DIN + 255) / 256, 256, 0, stream>>>(Bp_w,  alphas + 2, wq_Bp,  NST,  DIN, NSTP);
  k_wquant<<<(NSTP * DIN + 255) / 256, 256, 0, stream>>>(Cp_w,  alphas + 3, wq_Cp,  NST,  DIN, NSTP);
  k_wquant<<<(HDIM * DIN + 255) / 256, 256, 0, stream>>>(out_w, alphas + 4, wq_out, HDIM, DIN, HDIM);

  k_quant_in<<<TOK, 256, 0, stream>>>(x, norm_w, in_nw, xq_a, s_a);
  {
    int tiles = (TOK / 32) * (DXZ / 32);
    k_gemm_iu8<false><<<(tiles + 7) / 8, 256, 0, stream>>>(xq_a, wq_in, s_a, alphas + 0,
                                                           nullptr, xz, TOK, DXZ, DXZ, HDIM);
  }

  k_conv_silu<<<(TOK * DIN + 255) / 256, 256, 0, stream>>>(xz, conv_w, conv_b, xpath);

  k_quant_din3<<<TOK, 256, 0, stream>>>(xpath, dtu_nw, xq_dtu, s_dtu,
                                        Bp_nw, xq_Bp, s_Bp, Cp_nw, xq_Cp, s_Cp);
  {
    int tiles = (TOK / 32) * (DTRP2 / 32);
    k_gemm_iu8<false><<<(tiles + 7) / 8, 256, 0, stream>>>(xq_dtu, wq_dtu, s_dtu, alphas + 1,
                                                           nullptr, dtuout, TOK, DTRP2, DTRP, DIN);
  }
  {
    int tiles = (TOK / 32) * (NSTP / 32);
    k_gemm_iu8<false><<<(tiles + 7) / 8, 256, 0, stream>>>(xq_Bp, wq_Bp, s_Bp, alphas + 2,
                                                           nullptr, Bmat, TOK, NSTP, NST, DIN);
    k_gemm_iu8<false><<<(tiles + 7) / 8, 256, 0, stream>>>(xq_Cp, wq_Cp, s_Cp, alphas + 3,
                                                           nullptr, Cmat, TOK, NSTP, NST, DIN);
  }
  k_dtd<<<TOK, 256, 0, stream>>>(dtuout, dtd_w, dtd_b, dtbuf);

  dim3 gscan(DIN / 256, NCHUNK, NBATCH);
  k_scan_chunk<<<gscan, 256, 0, stream>>>(dtbuf, xpath, Bmat, A_log, Ach, Bch);
  k_scan_combine<<<(NBATCH * DIN + 255) / 256, 256, 0, stream>>>(Ach, Bch, hst);
  k_scan_apply<<<gscan, 256, 0, stream>>>(dtbuf, xpath, Bmat, Cmat, A_log, hst, ybuf);

  k_combine<<<(TOK * DIN + 255) / 256, 256, 0, stream>>>(xpath, Dvec, xz, ybuf);
  k_quant_din3<<<TOK, 256, 0, stream>>>(ybuf, out_nw, xq_y, s_y,
                                        nullptr, nullptr, nullptr, nullptr, nullptr, nullptr);
  {
    int tiles = (TOK / 32) * (HDIM / 32);
    k_gemm_iu8<true><<<(tiles + 7) / 8, 256, 0, stream>>>(xq_y, wq_out, s_y, alphas + 4,
                                                          x, out, TOK, HDIM, HDIM, DIN);
  }
}
